// DirectAssemble_17051020165265
// MI455X (gfx1250) — compile-verified, matrix operand order checked by reading, not run on hardware
//
#include <hip/hip_runtime.h>
#include <stdint.h>

// ---------------------------------------------------------------------------
// Types / helpers
// ---------------------------------------------------------------------------
typedef __attribute__((ext_vector_type(16))) __bf16 v16bf;
typedef __attribute__((ext_vector_type(8)))  float  v8f;

struct U8 { unsigned int v[8]; };   // 32 bytes == v16bf

static __device__ __forceinline__ unsigned short f2bf(float f) {
  unsigned int u = __builtin_bit_cast(unsigned int, f);
  u += 0x7FFFu + ((u >> 16) & 1u);          // round-to-nearest-even
  return (unsigned short)(u >> 16);
}

// Load one 16-bf16 fragment half-pair: 8 contiguous bf16 at p, 8 at p+16.
// p must already include the half*8 element offset (16B aligned).
static __device__ __forceinline__ U8 load_frag(const unsigned short* p) {
  uint4 lo = *(const uint4*)(p);
  uint4 hi = *(const uint4*)(p + 16);
  U8 r;
  r.v[0] = lo.x; r.v[1] = lo.y; r.v[2] = lo.z; r.v[3] = lo.w;
  r.v[4] = hi.x; r.v[5] = hi.y; r.v[6] = hi.z; r.v[7] = hi.w;
  return r;
}

#define NPIX 32768      // B*H*W = 8*64*64
#define NROWS 512       // B*H    = 8*64

// ---------------------------------------------------------------------------
// Pack: f32 NCHW (f || f_h) -> bf16 NHWC cat [NPIX][512]
// ---------------------------------------------------------------------------
__global__ void pack_cat_kernel(const float* __restrict__ f,
                                const float* __restrict__ fh,
                                unsigned short* __restrict__ cat) {
  int tid = blockIdx.x * blockDim.x + threadIdx.x;
  if (tid >= NPIX * 512) return;
  int n = tid >> 9, c = tid & 511;
  int b = n >> 12, rem = n & 4095;
  float v = (c < 256) ? f [(((size_t)b * 256 + c      ) << 12) + rem]
                      : fh[(((size_t)b * 256 + (c-256)) << 12) + rem];
  cat[tid] = f2bf(v);
}

// 1x1 conv weights [O][I] f32 -> bf16 (layout unchanged)
__global__ void pack_w_direct_kernel(const float* __restrict__ w,
                                     unsigned short* __restrict__ wp, int count) {
  int tid = blockIdx.x * blockDim.x + threadIdx.x;
  if (tid < count) wp[tid] = f2bf(w[tid]);
}

// 3x3 conv weights OIHW f32 -> [Opad][(kh*3+kw)*I + i] bf16, zero-pad o >= O
__global__ void pack_w3_kernel(const float* __restrict__ w,
                               unsigned short* __restrict__ wp,
                               int O, int Opad, int I) {
  int tid = blockIdx.x * blockDim.x + threadIdx.x;
  int total = Opad * 9 * I;
  if (tid >= total) return;
  int o   = tid / (9 * I);
  int r   = tid - o * 9 * I;
  int win = r / I;
  int i   = r - win * I;
  float v = (o < O) ? w[(size_t)(o * I + i) * 9 + win] : 0.0f;
  wp[tid] = f2bf(v);
}

// ---------------------------------------------------------------------------
// Implicit-GEMM conv via WMMA bf16 (wave32).
// Each wave computes a 16(M) x 64(N) tile: one weight (A) fragment is reused
// against 4 pixel sub-tiles (B fragments). blockIdx.x = image row (64 pixels),
// so b/y and dy-validity are wave-uniform for 3x3 convs.
// Activations: bf16 NHWC [NPIX][CIN]. Weights: bf16 [M][WIN*WIN*CIN] with
// k = (dy*WIN+dx)*CIN + c. Requires CIN % 32 == 0 so a K-step of 32 stays
// inside one window position.
// ---------------------------------------------------------------------------
template<int CIN, int WIN, bool RELU, bool OUTF32>
__global__ void conv_wmma_kernel(const unsigned short* __restrict__ xact,
                                 const unsigned short* __restrict__ wpack,
                                 void* __restrict__ outp,
                                 const float* __restrict__ bias,
                                 int mtiles, int mstride, int mstore) {
  constexpr int KTOT = CIN * WIN * WIN;
  const int wavesPerBlk = blockDim.x >> 5;
  const int wave  = threadIdx.x >> 5;
  const int lane  = threadIdx.x & 31;
  const int mtile = blockIdx.y * wavesPerBlk + wave;
  if (mtile >= mtiles) return;
  const int rowid = blockIdx.x;            // 0..511: b = rowid>>6, y = rowid&63
  const int b = rowid >> 6, y = rowid & 63;
  const int half = lane >> 4;
  const int hoff = half << 3;              // half * 8 elements
  const int mrow = lane & 15;
  const int ncol = lane & 15;

  v8f acc[4];
#pragma unroll
  for (int s = 0; s < 4; ++s) acc[s] = (v8f){0.f,0.f,0.f,0.f,0.f,0.f,0.f,0.f};

  const unsigned short* arow = wpack + (size_t)(mtile * 16 + mrow) * KTOT;

  for (int kb = 0; kb < KTOT; kb += 32) {
    // Prefetch weight stream ahead (global_prefetch_b8).
    __builtin_prefetch(arow + kb + 64, 0, 0);

    U8 au = load_frag(arow + kb + hoff);
    v16bf A = __builtin_bit_cast(v16bf, au);

    if (WIN == 1) {
#pragma unroll
      for (int sub = 0; sub < 4; ++sub) {
        const unsigned short* bp =
            xact + (size_t)(rowid * 64 + sub * 16 + ncol) * CIN + kb + hoff;
        U8 bu = load_frag(bp);
        v16bf Bf = __builtin_bit_cast(v16bf, bu);
        acc[sub] = __builtin_amdgcn_wmma_f32_16x16x32_bf16(
            false, A, false, Bf, (short)0, acc[sub], false, false);
      }
    } else {
      const int blk = kb / CIN;            // window position for this K-step
      const int dy = blk / 3, dx = blk - dy * 3;
      const int yy = y + dy - 1;
      const bool rowok = (unsigned)yy < 64u;   // wave-uniform
      const int c0 = kb - blk * CIN;
      const unsigned short* base =
          xact + (size_t)((b << 12) + yy * 64) * CIN + c0 + hoff;
#pragma unroll
      for (int sub = 0; sub < 4; ++sub) {
        const int xx = sub * 16 + ncol + dx - 1;
        const bool valid = rowok && ((unsigned)xx < 64u);
        U8 bu;
        if (valid) {
          bu = load_frag(base + (size_t)xx * CIN);
        } else {
#pragma unroll
          for (int i = 0; i < 8; ++i) bu.v[i] = 0u;
        }
        v16bf Bf = __builtin_bit_cast(v16bf, bu);
        acc[sub] = __builtin_amdgcn_wmma_f32_16x16x32_bf16(
            false, A, false, Bf, (short)0, acc[sub], false, false);
      }
    }
  }

  // Store D: VGPR r, lane -> M = r + 8*half, N = lane%16 (NHWC out)
#pragma unroll
  for (int sub = 0; sub < 4; ++sub) {
    const int pixel = rowid * 64 + sub * 16 + ncol;
#pragma unroll
    for (int r = 0; r < 8; ++r) {
      int m = mtile * 16 + r + hoff;
      if (m < mstore) {
        float v = acc[sub][r] + bias[m];
        if (RELU) v = v > 0.f ? v : 0.f;
        size_t idx = (size_t)pixel * mstride + m;
        if (OUTF32) ((float*)outp)[idx] = v;
        else        ((unsigned short*)outp)[idx] = f2bf(v);
      }
    }
  }
}

// ---------------------------------------------------------------------------
// assemble: out[n][c] = sum_{dy,dx in 9x9} atten[n][dy*9+dx] * f_h_pad(...)
// One block per image row (64 pixels), 256 threads: lanes sweep x so f_h row
// reads coalesce; 64x81 attention weights cached in LDS (20.25 KB).
// Writes aligned f32 NHWC and bf16 into upper half of cat (-> cat2).
// ---------------------------------------------------------------------------
__global__ void assemble_kernel(const float* __restrict__ atten81,
                                const float* __restrict__ fh,
                                float* __restrict__ aligned,
                                unsigned short* __restrict__ cat2) {
  const int rowid = blockIdx.x;            // 0..511
  const int b = rowid >> 6, y = rowid & 63;
  const int t = threadIdx.x;               // 0..255
  __shared__ float att[64 * 81];
  for (int i = t; i < 64 * 81; i += 256) {
    int px = i / 81, idx = i - px * 81;
    att[i] = atten81[(size_t)(rowid * 64 + px) * 81 + idx];
  }
  __syncthreads();
  const int x = t & 63;                    // pixel within row
  const int cbase = (t >> 6) << 6;         // 0,64,128,192
  const int n = rowid * 64 + x;
  const float* attx = &att[x * 81];
  for (int cc = 0; cc < 64; ++cc) {
    const int c = cbase + cc;
    const float* fhc = fh + (((size_t)b * 256 + c) << 12);
    float acc = 0.f;
    for (int dy = 0; dy < 9; ++dy) {
      int yy = y + dy - 4;
      if ((unsigned)yy >= 64u) continue;
      const float* rowp = fhc + yy * 64;
#pragma unroll
      for (int dx = 0; dx < 9; ++dx) {
        int xx = x + dx - 4;
        float v = ((unsigned)xx < 64u) ? rowp[xx] : 0.f;
        acc += attx[dy * 9 + dx] * v;
      }
    }
    aligned[(size_t)n * 256 + c] = acc;
    cat2[(size_t)n * 512 + 256 + c] = f2bf(acc);
  }
}

// ---------------------------------------------------------------------------
// Tail: g3 = conv3x3(g2[16ch] -> 3ch), scalar per pixel
// ---------------------------------------------------------------------------
__global__ void g3_kernel(const float* __restrict__ g2,
                          const float* __restrict__ fw3,
                          const float* __restrict__ fb3,
                          float* __restrict__ g3) {
  int n = blockIdx.x * blockDim.x + threadIdx.x;
  if (n >= NPIX) return;
  int b = n >> 12, rem = n & 4095, y = rem >> 6, x = rem & 63;
  float a0 = fb3[0], a1 = fb3[1], a2 = fb3[2];
  for (int dy = 0; dy < 3; ++dy) {
    int yy = y + dy - 1;
    if ((unsigned)yy >= 64u) continue;
    for (int dx = 0; dx < 3; ++dx) {
      int xx = x + dx - 1;
      if ((unsigned)xx >= 64u) continue;
      const float* gp = g2 + (size_t)((b << 12) + yy * 64 + xx) * 16;
      int widx = dy * 3 + dx;
#pragma unroll
      for (int i = 0; i < 16; ++i) {
        float v = gp[i];
        a0 += v * fw3[(size_t)(0 * 16 + i) * 9 + widx];
        a1 += v * fw3[(size_t)(1 * 16 + i) * 9 + widx];
        a2 += v * fw3[(size_t)(2 * 16 + i) * 9 + widx];
      }
    }
  }
  g3[(size_t)n * 3 + 0] = a0;
  g3[(size_t)n * 3 + 1] = a1;
  g3[(size_t)n * 3 + 2] = a2;
}

// logits = conv3x3(g3[3ch] -> 2ch) + softmax -> score[n][2]
__global__ void score_kernel(const float* __restrict__ g3,
                             const float* __restrict__ fw4,
                             const float* __restrict__ fb4,
                             float* __restrict__ score) {
  int n = blockIdx.x * blockDim.x + threadIdx.x;
  if (n >= NPIX) return;
  int b = n >> 12, rem = n & 4095, y = rem >> 6, x = rem & 63;
  float l0 = fb4[0], l1 = fb4[1];
  for (int dy = 0; dy < 3; ++dy) {
    int yy = y + dy - 1;
    if ((unsigned)yy >= 64u) continue;
    for (int dx = 0; dx < 3; ++dx) {
      int xx = x + dx - 1;
      if ((unsigned)xx >= 64u) continue;
      const float* gp = g3 + (size_t)((b << 12) + yy * 64 + xx) * 3;
      int widx = dy * 3 + dx;
#pragma unroll
      for (int i = 0; i < 3; ++i) {
        float v = gp[i];
        l0 += v * fw4[(size_t)(0 * 3 + i) * 9 + widx];
        l1 += v * fw4[(size_t)(1 * 3 + i) * 9 + widx];
      }
    }
  }
  float m  = fmaxf(l0, l1);
  float e0 = __expf(l0 - m), e1 = __expf(l1 - m);
  float inv = 1.0f / (e0 + e1);
  score[(size_t)n * 2 + 0] = e0 * inv;
  score[(size_t)n * 2 + 1] = e1 * inv;
}

// out(NCHW) = s0 * f + s1 * aligned
__global__ void blend_kernel(const float* __restrict__ f,
                             const float* __restrict__ aligned,
                             const float* __restrict__ score,
                             float* __restrict__ out) {
  int tid = blockIdx.x * blockDim.x + threadIdx.x;   // NCHW flat index
  if (tid >= NPIX * 256) return;
  int b   = tid >> 20;
  int c   = (tid >> 12) & 255;
  int rem = tid & 4095;
  int n   = (b << 12) | rem;
  float s0 = score[(size_t)n * 2 + 0];
  float s1 = score[(size_t)n * 2 + 1];
  out[tid] = s0 * f[tid] + s1 * aligned[(size_t)n * 256 + c];
}

// ---------------------------------------------------------------------------
// Workspace layout (bytes)
// ---------------------------------------------------------------------------
static constexpr size_t OFF_CAT     = 0;                                // 32768*512*2
static constexpr size_t OFF_A1      = OFF_CAT     + (size_t)NPIX*512*2; // 32768*256*2 (also g1)
static constexpr size_t OFF_A2      = OFF_A1      + (size_t)NPIX*256*2; // 32768*128*2
static constexpr size_t OFF_ATT     = OFF_A2      + (size_t)NPIX*128*2; // 32768*81*4
static constexpr size_t OFF_ALIGNED = OFF_ATT     + (size_t)NPIX*81*4;  // 32768*256*4
static constexpr size_t OFF_G2      = OFF_ALIGNED + (size_t)NPIX*256*4; // 32768*16*4
static constexpr size_t OFF_G3      = OFF_G2      + (size_t)NPIX*16*4;  // 32768*3*4
static constexpr size_t OFF_SCORE   = OFF_G3      + (size_t)NPIX*3*4;   // 32768*2*4
static constexpr size_t OFF_WA1     = OFF_SCORE   + (size_t)NPIX*2*4;   // 256*512*2
static constexpr size_t OFF_WA2     = OFF_WA1     + (size_t)256*512*2;  // 128*2304*2
static constexpr size_t OFF_WA3     = OFF_WA2     + (size_t)128*2304*2; // 96*1152*2
static constexpr size_t OFF_WF1     = OFF_WA3     + (size_t)96*1152*2;  // 256*512*2
static constexpr size_t OFF_WF2     = OFF_WF1     + (size_t)256*512*2;  // 16*2304*2

extern "C" void kernel_launch(void* const* d_in, const int* in_sizes, int n_in,
                              void* d_out, int out_size, void* d_ws, size_t ws_size,
                              hipStream_t stream) {
  (void)in_sizes; (void)n_in; (void)out_size; (void)ws_size;
  const float* f   = (const float*)d_in[0];
  const float* fh  = (const float*)d_in[1];
  const float* aw1 = (const float*)d_in[2];
  const float* ab1 = (const float*)d_in[3];
  const float* aw2 = (const float*)d_in[4];
  const float* ab2 = (const float*)d_in[5];
  const float* aw3 = (const float*)d_in[6];
  const float* ab3 = (const float*)d_in[7];
  const float* fw1 = (const float*)d_in[8];
  const float* fb1 = (const float*)d_in[9];
  const float* fw2 = (const float*)d_in[10];
  const float* fb2 = (const float*)d_in[11];
  const float* fw3 = (const float*)d_in[12];
  const float* fb3 = (const float*)d_in[13];
  const float* fw4 = (const float*)d_in[14];
  const float* fb4 = (const float*)d_in[15];
  float* out = (float*)d_out;

  char* ws = (char*)d_ws;
  unsigned short* cat     = (unsigned short*)(ws + OFF_CAT);
  unsigned short* a1      = (unsigned short*)(ws + OFF_A1);
  unsigned short* a2      = (unsigned short*)(ws + OFF_A2);
  float*          att     = (float*)(ws + OFF_ATT);
  float*          aligned = (float*)(ws + OFF_ALIGNED);
  float*          g2      = (float*)(ws + OFF_G2);
  float*          g3      = (float*)(ws + OFF_G3);
  float*          score   = (float*)(ws + OFF_SCORE);
  unsigned short* wa1     = (unsigned short*)(ws + OFF_WA1);
  unsigned short* wa2     = (unsigned short*)(ws + OFF_WA2);
  unsigned short* wa3     = (unsigned short*)(ws + OFF_WA3);
  unsigned short* wf1     = (unsigned short*)(ws + OFF_WF1);
  unsigned short* wf2     = (unsigned short*)(ws + OFF_WF2);

  // ---- pack activations and weights to bf16 ----
  pack_cat_kernel<<<(NPIX * 512 + 255) / 256, 256, 0, stream>>>(f, fh, cat);
  pack_w_direct_kernel<<<(256 * 512 + 255) / 256, 256, 0, stream>>>(aw1, wa1, 256 * 512);
  pack_w_direct_kernel<<<(256 * 512 + 255) / 256, 256, 0, stream>>>(fw1, wf1, 256 * 512);
  pack_w3_kernel<<<(128 * 9 * 256 + 255) / 256, 256, 0, stream>>>(aw2, wa2, 128, 128, 256);
  pack_w3_kernel<<<( 96 * 9 * 128 + 255) / 256, 256, 0, stream>>>(aw3, wa3,  81,  96, 128);
  pack_w3_kernel<<<( 16 * 9 * 256 + 255) / 256, 256, 0, stream>>>(fw2, wf2,  16,  16, 256);

  // ---- attention net (grid.x = 512 image rows, 64 pixels per row-tile) ----
  conv_wmma_kernel<512, 1, true,  false><<<dim3(NROWS, 4), 128, 0, stream>>>(
      cat, wa1, (void*)a1, ab1, 16, 256, 256);
  conv_wmma_kernel<256, 3, true,  false><<<dim3(NROWS, 2), 128, 0, stream>>>(
      a1, wa2, (void*)a2, ab2, 8, 128, 128);
  conv_wmma_kernel<128, 3, false, true ><<<dim3(NROWS, 2), 128, 0, stream>>>(
      a2, wa3, (void*)att, ab3, 6, 81, 81);

  // ---- assemble (also builds cat2 in-place in cat's upper channels) ----
  assemble_kernel<<<NROWS, 256, 0, stream>>>(att, fh, aligned, cat);

  // ---- final net ----
  conv_wmma_kernel<512, 1, true,  false><<<dim3(NROWS, 4), 128, 0, stream>>>(
      cat, wf1, (void*)a1, fb1, 16, 256, 256);           // g1 reuses a1 buffer
  conv_wmma_kernel<256, 3, true,  true ><<<dim3(NROWS, 1), 32, 0, stream>>>(
      a1, wf2, (void*)g2, fb2, 1, 16, 16);
  g3_kernel   <<<(NPIX + 255) / 256, 256, 0, stream>>>(g2, fw3, fb3, g3);
  score_kernel<<<(NPIX + 255) / 256, 256, 0, stream>>>(g3, fw4, fb4, score);
  blend_kernel<<<(NPIX * 256 + 255) / 256, 256, 0, stream>>>(f, aligned, score, out);
}
